// NetworkPRC_21079699489070
// MI455X (gfx1250) — matrix operand order repaired and checked by reading, not compiled, not run on hardware
//
#include <hip/hip_runtime.h>
#include <hip/hip_bf16.h>

// ---------------------------------------------------------------------------
// Problem constants
// ---------------------------------------------------------------------------
#define B_    16384
#define IN_   151
#define HID_  200
#define CENT_ 2000
// padded dims: K multiples of 32, N multiples of 64 (so a wave's 4 N-subtiles
// are always valid -> zero control flow in the WMMA loop)
#define KP1_  224     // HID 200 -> 224   (K of GEMM2)
#define NP_   2048    // CENT 2000 -> 2048 (N of GEMM2/3, K of GEMM3/4)
#define NP3_  192     // IN 151 -> 192    (N of GEMM4)

typedef __attribute__((ext_vector_type(16))) __bf16      v16bf;
typedef __attribute__((ext_vector_type(8)))  float       v8f;
typedef __attribute__((ext_vector_type(4)))  unsigned int v4u;

union Frag { v4u u[2]; v16bf v; };

__device__ __forceinline__ __bf16 f2bf(float f) {
    union { float f; unsigned u; } in; in.f = f;
    unsigned u = in.u;
    u += 0x7FFFu + ((u >> 16) & 1u);           // round-to-nearest-even
    union { unsigned short s; __bf16 b; } out;
    out.s = (unsigned short)(u >> 16);
    return out.b;
}

// ---------------------------------------------------------------------------
// f32 -> bf16 with 2-D zero padding (weights / kern conversion)
// ---------------------------------------------------------------------------
__global__ void pad_bf16(const float* __restrict__ src, __bf16* __restrict__ dst,
                         int rows, int cols, int rowsPad, int colsPad) {
    size_t idx = (size_t)blockIdx.x * blockDim.x + threadIdx.x;
    size_t total = (size_t)rowsPad * colsPad;
    if (idx >= total) return;
    int r = (int)(idx / colsPad);
    int c = (int)(idx % colsPad);
    float v = (r < rows && c < cols) ? src[(size_t)r * cols + c] : 0.0f;
    dst[idx] = f2bf(v);
}

// ---------------------------------------------------------------------------
// RBF layer: phi[i,j] = exp(-||x_i - c_j||^2 * sigma_j^2), bf16, K-padded
// ---------------------------------------------------------------------------
__global__ __launch_bounds__(256)
void rbf_phi(const float* __restrict__ x, const float* __restrict__ centres,
             const float* __restrict__ sigmas, __bf16* __restrict__ phiB) {
    __shared__ float xs[IN_];
    const int i = blockIdx.x;
    for (int k = threadIdx.x; k < IN_; k += blockDim.x)
        xs[k] = x[(size_t)i * IN_ + k];
    __syncthreads();
    const int j = threadIdx.x;
    if (j < KP1_) {
        float v = 0.0f;
        if (j < HID_) {
            const float* c = centres + (size_t)j * IN_;
            float acc = 0.0f;
            #pragma unroll 4
            for (int k = 0; k < IN_; ++k) {
                float d = xs[k] - c[k];
                acc = fmaf(d, d, acc);
            }
            float s = sigmas[j];
            v = __expf(-acc * s * s);
        }
        phiB[(size_t)i * KP1_ + j] = f2bf(v);
    }
}

// ---------------------------------------------------------------------------
// bf16 WMMA GEMM:  C[M x Np] = epi( A[M x Kp] @ W[Np x Kp]^T + bias )
// one wave computes a 32x64 strip: 2 A-frags x 4 B-frags = 8 WMMAs / K-step.
// Grid is sized exactly (jobs % waves-per-block == 0): no idle waves, no
// divergent control flow anywhere near the WMMAs -> EXEC stays all-ones.
// ---------------------------------------------------------------------------
#define EPI_NONE 0
#define EPI_RELU 1
#define EPI_EXP  2

template<int EPI>
__global__ __launch_bounds__(256)
void wmma_gemm_bf16(const __bf16* __restrict__ A, int lda,
                    const __bf16* __restrict__ W, int ldw,
                    const float* __restrict__ bias, int nBias,
                    float* __restrict__ outF, int ldf,
                    __bf16* __restrict__ outB, int ldb,
                    int mtiles32, int Kp, int Nreal) {
    const int lane = threadIdx.x & 31;
    const int half = lane >> 4;            // ISA 16-bit operand lane split
    const int lm   = lane & 15;
    const int wave = blockIdx.x * (blockDim.x >> 5) + (threadIdx.x >> 5);
    const int mt   = wave % mtiles32;      // 32-row strip index
    const int ng   = wave / mtiles32;      // 64-col group index

    v8f acc0[4] = {v8f{}, v8f{}, v8f{}, v8f{}};
    v8f acc1[4] = {v8f{}, v8f{}, v8f{}, v8f{}};

    // A layout (ISA 7.12.2, 16-bit 16x32): lane(m,half) holds
    // K in [k0+8h, +8) and [k0+16+8h, +8)
    const __bf16* arow0 = A + (size_t)(mt * 32 + lm) * lda + half * 8;
    const __bf16* arow1 = arow0 + (size_t)16 * lda;
    // B: lane holds column n = base+lm, K in [k0+8h, +8) and [k0+16+8h, +8)
    const __bf16* wbase = W + (size_t)(ng * 64 + lm) * ldw + half * 8;

    for (int k0 = 0; k0 < Kp; k0 += 32) {
        Frag a0, a1;
        a0.u[0] = *(const v4u*)(arow0 + k0);
        a0.u[1] = *(const v4u*)(arow0 + k0 + 16);
        a1.u[0] = *(const v4u*)(arow1 + k0);
        a1.u[1] = *(const v4u*)(arow1 + k0 + 16);
        #pragma unroll
        for (int t = 0; t < 4; ++t) {
            const __bf16* wrow = wbase + (size_t)(t * 16) * ldw + k0;
            Frag b;
            b.u[0] = *(const v4u*)(wrow);
            b.u[1] = *(const v4u*)(wrow + 16);
            acc0[t] = __builtin_amdgcn_wmma_f32_16x16x32_bf16(
                          false, a0.v, false, b.v, (short)0, acc0[t], false, false);
            acc1[t] = __builtin_amdgcn_wmma_f32_16x16x32_bf16(
                          false, a1.v, false, b.v, (short)0, acc1[t], false, false);
        }
    }

    #pragma unroll
    for (int t = 0; t < 4; ++t) {
        const int ncol = ng * 64 + t * 16 + lm;
        float bv = 0.0f;
        if (EPI != EPI_NONE)
            bv = (bias != nullptr && ncol < nBias) ? bias[ncol] : 0.0f;
        #pragma unroll
        for (int half2 = 0; half2 < 2; ++half2) {
            #pragma unroll
            for (int v = 0; v < 8; ++v) {
                // C layout: lanes 0-15 VGPR v -> M=v; lanes 16-31 -> M=v+8
                const int row = mt * 32 + half2 * 16 + v + 8 * half;
                float val = (half2 ? acc1[t][v] : acc0[t][v]) + bv;
                if (EPI == EPI_NONE) {
                    if (ncol < Nreal) outF[(size_t)row * ldf + ncol] = val;
                } else if (EPI == EPI_RELU) {
                    float r = val > 0.0f ? val : 0.0f;
                    outB[(size_t)row * ldb + ncol] = f2bf(ncol < Nreal ? r : 0.0f);
                } else { // EPI_EXP
                    float e = __expf(val);
                    if (ncol < Nreal) outF[(size_t)row * ldf + ncol] = e;
                    outB[(size_t)row * ldb + ncol] = f2bf(ncol < Nreal ? e : 0.0f);
                }
            }
        }
    }
}

// ---------------------------------------------------------------------------
// Finalize: e0 = Ehat[i,0];  Ehat2 = Ehat/e0 ;  Rhat = out/e0 ;  copy x
// Outputs concatenated: [Ehat2, Rhat, Ehat2, Rhat, x]
// ---------------------------------------------------------------------------
__global__ __launch_bounds__(256)
void finalize(const float* __restrict__ Ehat,   // B x NP3_ (ld NP3_)
              const float* __restrict__ outF,   // B x CENT_
              const float* __restrict__ x,      // B x IN_
              float* __restrict__ out) {
    const size_t o0 = 0;
    const size_t o1 = (size_t)B_ * IN_;
    const size_t o2 = o1 + (size_t)B_ * CENT_;
    const size_t o3 = o2 + (size_t)B_ * IN_;
    const size_t o4 = o3 + (size_t)B_ * CENT_;

    const int i = blockIdx.x;
    const float inv = 1.0f / Ehat[(size_t)i * NP3_];

    for (int j = threadIdx.x; j < IN_; j += blockDim.x) {
        float e2 = Ehat[(size_t)i * NP3_ + j] * inv;
        out[o0 + (size_t)i * IN_ + j] = e2;
        out[o2 + (size_t)i * IN_ + j] = e2;
        out[o4 + (size_t)i * IN_ + j] = x[(size_t)i * IN_ + j];
    }
    for (int c = threadIdx.x; c < CENT_; c += blockDim.x) {
        float r = outF[(size_t)i * CENT_ + c] * inv;
        out[o1 + (size_t)i * CENT_ + c] = r;
        out[o3 + (size_t)i * CENT_ + c] = r;
    }
}

// ---------------------------------------------------------------------------
// Launch
// ---------------------------------------------------------------------------
extern "C" void kernel_launch(void* const* d_in, const int* in_sizes, int n_in,
                              void* d_out, int out_size, void* d_ws, size_t ws_size,
                              hipStream_t stream) {
    (void)in_sizes; (void)n_in; (void)out_size; (void)ws_size;
    const float* x       = (const float*)d_in[0];
    const float* centres = (const float*)d_in[1];
    const float* sigmas  = (const float*)d_in[2];
    const float* w1      = (const float*)d_in[3];
    const float* b1      = (const float*)d_in[4];
    const float* w2      = (const float*)d_in[5];
    const float* b2      = (const float*)d_in[6];
    const float* kern    = (const float*)d_in[7];
    float* out = (float*)d_out;

    // workspace carve (256B aligned)
    char* p = (char*)d_ws;
    auto carve = [&](size_t bytes) {
        char* r = p; p += (bytes + 255) & ~(size_t)255; return (void*)r;
    };
    __bf16* phiB  = (__bf16*)carve((size_t)B_   * KP1_ * 2);
    __bf16* w1B   = (__bf16*)carve((size_t)NP_  * KP1_ * 2);
    __bf16* hB    = (__bf16*)carve((size_t)B_   * NP_  * 2);
    __bf16* w2B   = (__bf16*)carve((size_t)NP_  * NP_  * 2);
    float*  outF  = (float*) carve((size_t)B_   * CENT_ * 4);
    __bf16* outB  = (__bf16*)carve((size_t)B_   * NP_  * 2);
    __bf16* kernB = (__bf16*)carve((size_t)NP3_ * NP_  * 2);
    float*  EhatF = (float*) carve((size_t)B_   * NP3_ * 4);

    // 1) weight conversions (f32 -> bf16, zero-padded)
    {
        size_t t1 = (size_t)NP_  * KP1_;
        size_t t2 = (size_t)NP_  * NP_;
        size_t t3 = (size_t)NP3_ * NP_;
        pad_bf16<<<(unsigned)((t1 + 255) / 256), 256, 0, stream>>>(w1,   w1B,   CENT_, HID_,  NP_,  KP1_);
        pad_bf16<<<(unsigned)((t2 + 255) / 256), 256, 0, stream>>>(w2,   w2B,   CENT_, CENT_, NP_,  NP_);
        pad_bf16<<<(unsigned)((t3 + 255) / 256), 256, 0, stream>>>(kern, kernB, IN_,   CENT_, NP3_, NP_);
    }

    // 2) RBF layer -> phi (bf16, B x 224)
    rbf_phi<<<B_, 256, 0, stream>>>(x, centres, sigmas, phiB);

    // jobs = (B/32) * (Np/64); all divisible by 8 waves/block -> exact grids
    const int mtiles32 = B_ / 32;                       // 512
    const unsigned blk_main = (unsigned)(mtiles32 * (NP_  / 64) / 8);  // 2048
    const unsigned blk_ehat = (unsigned)(mtiles32 * (NP3_ / 64) / 8);  // 192

    // 3) h = relu(phi @ w1^T + b1)      (B x 2048 bf16, pad cols zero)
    wmma_gemm_bf16<EPI_RELU><<<blk_main, 256, 0, stream>>>(
        phiB, KP1_, w1B, KP1_, b1, CENT_,
        nullptr, 0, hB, NP_, mtiles32, KP1_, CENT_);

    // 4) out = exp(h @ w2^T + b2)       (f32 B x 2000, bf16 B x 2048)
    wmma_gemm_bf16<EPI_EXP><<<blk_main, 256, 0, stream>>>(
        hB, NP_, w2B, NP_, b2, CENT_,
        outF, CENT_, outB, NP_, mtiles32, NP_, CENT_);

    // 5) Ehat = out @ kern^T            (f32 B x 192; cols >=151 are zero)
    wmma_gemm_bf16<EPI_NONE><<<blk_ehat, 256, 0, stream>>>(
        outB, NP_, kernB, NP_, nullptr, 0,
        EhatF, NP3_, nullptr, 0, mtiles32, NP_, NP3_);

    // 6) Ehat2 = Ehat/e0, Rhat = out/e0, assemble outputs
    finalize<<<B_, 256, 0, stream>>>(EhatF, outF, x, out);
}
